// PatchMambaVectorField_71768903516206
// MI455X (gfx1250) — matrix operand order repaired and checked by reading, not compiled, not run on hardware
//
#include <hip/hip_runtime.h>
#include <hip/hip_bf16.h>
#include <math.h>

// ---------------------------------------------------------------------------
// PatchMambaVectorField for MI455X (gfx1250, wave32, WMMA)
//   - every matmul (K % 32 == 0) runs through v_wmma_f32_16x16x32_f16
//   - tile loads: unconditional float4 -> global_load_b128, fp16 LDS staging
//   - K=8 delta projection handled by a dedicated fused softplus kernel
//   - selective scan keeps h[256x64] per batch in registers (32 cells/lane)
// ---------------------------------------------------------------------------

typedef __attribute__((ext_vector_type(16))) _Float16 v16h;
typedef __attribute__((ext_vector_type(4)))  _Float16 v4h;
typedef __attribute__((ext_vector_type(8)))  float    v8f;

#define B_    32
#define P_    32
#define L_    80               // NPATCH
#define DM_   128
#define DS_   64
#define DC_   4
#define DI_   256
#define DTR_  8
#define MROWS (B_ * L_)        // 2560
#define XDBL_ (DTR_ + 2 * DS_) // 136

__device__ __forceinline__ int rmap(int r, int rev) {
  // per-L_-chunk row reversal; L_ is compile-time -> mul/shift, no int div
  return rev ? ((r / L_) * L_ + (L_ - 1 - (r % L_))) : r;
}
__device__ __forceinline__ float sigmoidf_(float x) { return 1.f / (1.f + __expf(-x)); }

// ---------------------------------------------------------------------------
// WMMA GEMM: out[m,n] = epi( sum_k A[m,k] * W[n,k] + bias[n] )
//   REQUIRES: M % 32 == 0 and K % 32 == 0 (true for all launches here).
//   A row stride = lda; optional per-L_-chunk row reversal on A read (revA)
//   and on C write (revC).  epi: 0 none, 2 add addvec[(m/L_)*N+n].
// Block = 256 threads = 8 waves; block tile 32(M) x 64(N); K-step 32.
// ---------------------------------------------------------------------------
__global__ __launch_bounds__(256) void gemm_wmma_f16(
    const float* __restrict__ A, int lda,
    const float* __restrict__ W,
    const float* __restrict__ bias,
    float* __restrict__ out, int ldc,
    int M, int N, int K,
    int revA, int revC, int epi,
    const float* __restrict__ addvec)
{
  __shared__ _Float16 Atile[32][40];   // [m][k], padded (pitch 80 B)
  __shared__ _Float16 Btile[64][40];   // [n][k], padded (k contiguous)

  const int tid  = threadIdx.x;
  const int lane = tid & 31;
  const int wave = tid >> 5;
  const int wy = wave >> 2;            // 0..1 -> M sub-tile
  const int wx = wave & 3;             // 0..3 -> N sub-tile
  const int tileM = blockIdx.y * 32;
  const int tileN = blockIdx.x * 64;

  // --- per-thread load coordinates (invariant over K loop)
  const int arow  = tid >> 3;              // 0..31
  const int akq   = (tid & 7) << 2;        // 0,4,...,28
  const int agrow = rmap(tileM + arow, revA);
  const float* __restrict__ abase = A + (long)agrow * lda + akq;

  const int bn  = tid >> 2;                // 0..63
  const int bkq = (tid & 3) << 3;          // 0,8,16,24
  const int gnb = tileN + bn;
  const int gnc = (gnb < N) ? gnb : 0;     // clamped -> loads stay in-bounds
  const float bmask = (gnb < N) ? 1.f : 0.f;
  const float* __restrict__ wbase = W + (long)gnc * K + bkq;

  v8f acc = {};

  for (int k0 = 0; k0 < K; k0 += 32) {
    // --- A tile: one b128 load + packed cvt + one b64 LDS store
    {
      float4 v = *reinterpret_cast<const float4*>(abase + k0);
      v4h h4 = { (_Float16)v.x, (_Float16)v.y, (_Float16)v.z, (_Float16)v.w };
      *reinterpret_cast<v4h*>(&Atile[arow][akq]) = h4;
    }
    // --- B tile: two b128 loads, mask-multiplied (branchless N edge)
    {
      float4 a = *reinterpret_cast<const float4*>(wbase + k0);
      float4 b = *reinterpret_cast<const float4*>(wbase + k0 + 4);
      v4h h0 = { (_Float16)(a.x * bmask), (_Float16)(a.y * bmask),
                 (_Float16)(a.z * bmask), (_Float16)(a.w * bmask) };
      v4h h1 = { (_Float16)(b.x * bmask), (_Float16)(b.y * bmask),
                 (_Float16)(b.z * bmask), (_Float16)(b.w * bmask) };
      *reinterpret_cast<v4h*>(&Btile[bn][bkq])     = h0;
      *reinterpret_cast<v4h*>(&Btile[bn][bkq + 4]) = h1;
    }
    __syncthreads();

    // --- fragment gather per ISA 16-bit A/B layouts (wave32)
    const int m16 = lane & 15;
    const int hl  = lane >> 4;
    v16h afrag, bfrag;
    const _Float16* ar = &Atile[wy * 16 + m16][0];
    #pragma unroll
    for (int v = 0; v < 8; ++v) {
      int k = ((v >> 2) << 4) + (hl << 3) + ((v & 3) << 1);  // contiguous K pairs
      afrag[2 * v]     = ar[k];
      afrag[2 * v + 1] = ar[k + 1];
    }
    const _Float16* br = &Btile[wx * 16 + m16][0];
    #pragma unroll
    for (int v = 0; v < 8; ++v) {
      int k = ((v >> 2) << 4) + (hl << 3) + ((v & 3) << 1);
      bfrag[2 * v]     = br[k];
      bfrag[2 * v + 1] = br[k + 1];
    }

    acc = __builtin_amdgcn_wmma_f32_16x16x32_f16(false, afrag, false, bfrag,
                                                 (short)0, acc, false, false);
    __syncthreads();
  }

  // --- epilogue + store (C/D layout: n = lane%16, m = r + 8*(lane/16))
  const int laneN = lane & 15;
  const int laneH = lane >> 4;
  const int gn = tileN + wx * 16 + laneN;
  if (gn < N) {
    float bv = bias ? bias[gn] : 0.f;
    #pragma unroll
    for (int r = 0; r < 8; ++r) {
      int gm = tileM + wy * 16 + r + 8 * laneH;     // always < M (M % 32 == 0)
      float v = acc[r] + bv;
      if (epi == 2) v += addvec[(gm / L_) * N + gn];           // +temb[b]
      out[(long)rmap(gm, revC) * ldc + gn] = v;
    }
  }
}

// ---------------------------------------------------------------------------
// delta = softplus(xdbl[:, :8] @ W_dt^T + b_dt)   (K=8: VALU kernel, one
// block per row; dt vector staged in LDS, weights via two float4 loads)
// ---------------------------------------------------------------------------
__global__ __launch_bounds__(256) void delta_softplus_kernel(
    const float* __restrict__ xdbl,   // [M,136], dt in cols 0..7
    const float* __restrict__ W_dt,   // [DI,8]
    const float* __restrict__ b_dt,   // [DI]
    float* __restrict__ delta)        // [M,DI]
{
  const int m = blockIdx.x;
  const int n = threadIdx.x;
  __shared__ float dt[DTR_];
  if (n < DTR_) dt[n] = xdbl[(long)m * XDBL_ + n];
  __syncthreads();
  const float4 w0 = *reinterpret_cast<const float4*>(W_dt + n * DTR_);
  const float4 w1 = *reinterpret_cast<const float4*>(W_dt + n * DTR_ + 4);
  float v = b_dt[n]
          + dt[0] * w0.x + dt[1] * w0.y + dt[2] * w0.z + dt[3] * w0.w
          + dt[4] * w1.x + dt[5] * w1.y + dt[6] * w1.z + dt[7] * w1.w;
  delta[(long)m * DI_ + n] = (v > 20.f) ? v : log1pf(__expf(v));
}

// ---------------------------------------------------------------------------
// temb = silu(t*W1 + b1) @ W2^T + b2   (tiny: one block per batch)
// ---------------------------------------------------------------------------
__global__ void time_mlp_kernel(const float* __restrict__ t,
                                const float* __restrict__ W1, const float* __restrict__ b1,
                                const float* __restrict__ W2, const float* __restrict__ b2,
                                float* __restrict__ temb)
{
  int b = blockIdx.x, j = threadIdx.x;
  __shared__ float s[DM_];
  float x = t[b] * W1[j] + b1[j];
  s[j] = x * sigmoidf_(x);
  __syncthreads();
  float acc = b2[j];
  for (int i = 0; i < DM_; ++i) acc += s[i] * W2[j * DM_ + i];
  temb[b * DM_ + j] = acc;
}

// ---------------------------------------------------------------------------
// causal depthwise conv1d (DC=4) + bias + silu.  xz cols [0,256) hold xin_raw.
// Branchless: clamped index + mask multiply.
// ---------------------------------------------------------------------------
__global__ void conv_silu_kernel(const float* __restrict__ xz,
                                 const float* __restrict__ cw,   // [DI,DC]
                                 const float* __restrict__ cb,   // [DI]
                                 float* __restrict__ xin)        // [M,DI]
{
  int idx = blockIdx.x * blockDim.x + threadIdx.x;
  if (idx >= MROWS * DI_) return;
  int d = idx & (DI_ - 1);
  int r = idx >> 8;
  int l = r % L_;
  int b = r / L_;
  float s = cb[d];
  #pragma unroll
  for (int k = 0; k < DC_; ++k) {
    int ll = l - (DC_ - 1) + k;
    int lc = (ll >= 0) ? ll : 0;
    float m = (ll >= 0) ? 1.f : 0.f;
    s += m * cw[d * DC_ + k] * xz[(long)(b * L_ + lc) * (2 * DI_) + d];
  }
  xin[(long)r * DI_ + d] = s * sigmoidf_(s);
}

// ---------------------------------------------------------------------------
// selective scan, fused with +u*D skip and *silu(z) gate.
// One block per batch; lane pair (d, n-half) owns 32 state cells in registers.
// ---------------------------------------------------------------------------
__global__ __launch_bounds__(512) void scan_kernel(
    const float* __restrict__ xin,    // [M,DI] u
    const float* __restrict__ delta,  // [M,DI]
    const float* __restrict__ xdbl,   // [M,136] (Bs @ +8, Cs @ +72)
    const float* __restrict__ xz,     // [M,512] (z @ cols 256..)
    const float* __restrict__ A_log,  // [DI,DS]
    const float* __restrict__ Dp,     // [DI]
    float* __restrict__ ybuf)         // [M,DI]
{
  const int b = blockIdx.x;
  const int tid = threadIdx.x;
  const int d = tid >> 1;
  const int half = tid & 1;
  const int nb = half << 5;

  float h[32], a[32];
  #pragma unroll
  for (int j = 0; j < 32; ++j) {
    h[j] = 0.f;
    a[j] = -__expf(A_log[d * DS_ + nb + j]);
  }

  __shared__ float sB[DS_], sC[DS_], sDelta[DI_], sU[DI_], sZ[DI_];

  for (int l = 0; l < L_; ++l) {
    long row = (long)b * L_ + l;
    if (tid < 256) {
      sDelta[tid] = delta[row * DI_ + tid];
      sU[tid]     = xin[row * DI_ + tid];
    } else {
      int q = tid - 256;
      sZ[q] = xz[row * (2 * DI_) + DI_ + q];
      if (q < 64)       sB[q]      = xdbl[row * XDBL_ + DTR_ + q];
      else if (q < 128) sC[q - 64] = xdbl[row * XDBL_ + DTR_ + DS_ + (q - 64)];
    }
    __syncthreads();

    float dlt = sDelta[d];
    float du  = dlt * sU[d];
    float ps = 0.f;
    #pragma unroll
    for (int j = 0; j < 32; ++j) {
      float hj = __expf(dlt * a[j]) * h[j] + du * sB[nb + j];
      h[j] = hj;
      ps += hj * sC[nb + j];
    }
    ps += __shfl_xor(ps, 1, 32);      // combine the two n-halves of d
    if (half == 0) {
      float y = ps + sU[d] * Dp[d];
      float z = sZ[d];
      ybuf[row * DI_ + d] = y * (z * sigmoidf_(z));
    }
    __syncthreads();
  }
}

// ---------------------------------------------------------------------------
// lat = layernorm(lat + fused) * g + beta    (one block per row)
// ---------------------------------------------------------------------------
__global__ void ln_residual_kernel(float* __restrict__ lat,
                                   const float* __restrict__ fused,
                                   const float* __restrict__ g,
                                   const float* __restrict__ beta)
{
  int row = blockIdx.x, j = threadIdx.x;
  __shared__ float red[DM_];
  float v = lat[(long)row * DM_ + j] + fused[(long)row * DM_ + j];
  red[j] = v; __syncthreads();
  for (int s = 64; s > 0; s >>= 1) { if (j < s) red[j] += red[j + s]; __syncthreads(); }
  float mean = red[0] * (1.f / DM_);
  __syncthreads();
  float dv = v - mean;
  red[j] = dv * dv; __syncthreads();
  for (int s = 64; s > 0; s >>= 1) { if (j < s) red[j] += red[j + s]; __syncthreads(); }
  float var = red[0] * (1.f / DM_);
  lat[(long)row * DM_ + j] = dv * rsqrtf(var + 1e-5f) * g[j] + beta[j];
}

// ---------------------------------------------------------------------------
// Host driver.  Input index map (recursive insertion order of setup_inputs):
//   0 x_t, 1 t, 2 Wp, 3 bp, 4 W1, 5 b1, 6 W2, 7 b2,
//   block li (base = 8 + 22*li):
//     fwd mamba @ base+0..8 : W_in, conv_w, conv_b, W_x, W_dt, b_dt, A_log, Dp, W_out
//     bwd mamba @ base+9..17
//     Wf base+18, bf base+19, g base+20, beta base+21
//   74 Wh, 75 bh
// ---------------------------------------------------------------------------
extern "C" void kernel_launch(void* const* d_in, const int* in_sizes, int n_in,
                              void* d_out, int out_size, void* d_ws, size_t ws_size,
                              hipStream_t stream)
{
  (void)in_sizes; (void)n_in; (void)out_size; (void)ws_size;
  const float* x_t = (const float*)d_in[0];
  const float* t   = (const float*)d_in[1];
  const float* Wp  = (const float*)d_in[2];
  const float* bp  = (const float*)d_in[3];
  const float* W1  = (const float*)d_in[4];
  const float* b1  = (const float*)d_in[5];
  const float* W2  = (const float*)d_in[6];
  const float* b2  = (const float*)d_in[7];
  const float* Wh  = (const float*)d_in[74];
  const float* bh  = (const float*)d_in[75];

  float* ws = (float*)d_ws;
  float* lat   = ws; ws += (long)MROWS * DM_;
  float* temb  = ws; ws += (long)B_ * DM_;
  float* xz    = ws; ws += (long)MROWS * 2 * DI_;
  float* xin   = ws; ws += (long)MROWS * DI_;
  float* xdbl  = ws; ws += (long)MROWS * XDBL_;
  float* delta = ws; ws += (long)MROWS * DI_;
  float* ybuf  = ws; ws += (long)MROWS * DI_;
  float* cat   = ws; ws += (long)MROWS * 2 * DM_;
  float* fused = ws; ws += (long)MROWS * DM_;

  auto gemm = [&](const float* A, int lda, const float* W, const float* bias,
                  float* out, int ldc, int M, int N, int K,
                  int revA, int revC, int epi, const float* addvec) {
    dim3 grid((N + 63) / 64, (M + 31) / 32);
    gemm_wmma_f16<<<grid, 256, 0, stream>>>(A, lda, W, bias, out, ldc,
                                            M, N, K, revA, revC, epi, addvec);
  };

  time_mlp_kernel<<<B_, DM_, 0, stream>>>(t, W1, b1, W2, b2, temb);
  // lat = patches @ Wp^T + bp + temb[b]
  gemm(x_t, P_, Wp, bp, lat, DM_, MROWS, DM_, P_, 0, 0, 2, temb);

  for (int li = 0; li < 3; ++li) {
    int base = 8 + 22 * li;
    for (int dir = 0; dir < 2; ++dir) {      // 0 = fwd, 1 = bwd (flipped domain)
      int mb = base + 9 * dir;
      const float* W_in  = (const float*)d_in[mb + 0];
      const float* cw    = (const float*)d_in[mb + 1];
      const float* cb    = (const float*)d_in[mb + 2];
      const float* W_x   = (const float*)d_in[mb + 3];
      const float* W_dt  = (const float*)d_in[mb + 4];
      const float* b_dt  = (const float*)d_in[mb + 5];
      const float* A_log = (const float*)d_in[mb + 6];
      const float* Dp    = (const float*)d_in[mb + 7];
      const float* W_out = (const float*)d_in[mb + 8];

      // xz = (rev?flip(lat):lat) @ W_in^T            [M,512]
      gemm(lat, DM_, W_in, nullptr, xz, 2 * DI_, MROWS, 2 * DI_, DM_, dir, 0, 0, nullptr);
      // xin = silu(causal_dwconv(xz[:, :256]) + cb)  [M,256]
      conv_silu_kernel<<<(MROWS * DI_) / 256, 256, 0, stream>>>(xz, cw, cb, xin);
      // xdbl = xin @ W_x^T                           [M,136]
      gemm(xin, DI_, W_x, nullptr, xdbl, XDBL_, MROWS, XDBL_, DI_, 0, 0, 0, nullptr);
      // delta = softplus(xdbl[:, :8] @ W_dt^T + b_dt)[M,256]   (K=8 -> VALU kernel)
      delta_softplus_kernel<<<MROWS, DI_, 0, stream>>>(xdbl, W_dt, b_dt, delta);
      // ybuf = scan(...) * silu(z)                   [M,256]
      scan_kernel<<<B_, 512, 0, stream>>>(xin, delta, xdbl, xz, A_log, Dp, ybuf);
      // cat[:, dir*128 : dir*128+128] = (rev?flip:id)(ybuf @ W_out^T)
      gemm(ybuf, DI_, W_out, nullptr, cat + dir * DM_, 2 * DM_, MROWS, DM_, DI_,
           0, dir, 0, nullptr);
    }
    const float* Wf   = (const float*)d_in[base + 18];
    const float* bf   = (const float*)d_in[base + 19];
    const float* g    = (const float*)d_in[base + 20];
    const float* beta = (const float*)d_in[base + 21];
    gemm(cat, 2 * DM_, Wf, bf, fused, DM_, MROWS, DM_, 2 * DM_, 0, 0, 0, nullptr);
    ln_residual_kernel<<<MROWS, DM_, 0, stream>>>(lat, fused, g, beta);
  }

  // d_out = lat @ Wh^T + bh   -> [B, RAW, 1]
  gemm(lat, DM_, Wh, bh, (float*)d_out, P_, MROWS, P_, DM_, 0, 0, 0, nullptr);
}
